// TransformerBlock_37666863186272
// MI455X (gfx1250) — matrix operand order repaired and checked by reading, not compiled, not run on hardware
//
#include <hip/hip_runtime.h>

// ---------------------------------------------------------------------------
// CDNA5 (gfx1250) transformer block, bf16 WMMA (v_wmma_f32_16x16x32_bf16),
// fp32 accumulate. ~450 GFLOP vs ~25us of HBM traffic at 23.3 TB/s ->
// matrix-compute bound, so all GEMM-like work goes through the WMMA pipe.
// ---------------------------------------------------------------------------

typedef unsigned short u16;
typedef unsigned int   u32;

typedef __attribute__((ext_vector_type(16))) __bf16 v16bf;
typedef __attribute__((ext_vector_type(8)))  float  v8f;
typedef __attribute__((ext_vector_type(4)))  u32    u32x4;

union Frag {
  u32x4 q[2];
  v16bf v;
  u16   u[16];
};

__device__ __forceinline__ u16 f2bf(float f) {
  u32 u = __float_as_uint(f);
  u += 0x7FFFu + ((u >> 16) & 1u);      // round-to-nearest-even
  return (u16)(u >> 16);
}

__device__ __forceinline__ float gelu_tanh(float x) {
  return 0.5f * x * (1.0f + tanhf(0.7978845608028654f * (x + 0.044715f * x * x * x)));
}

// A-fragment (16x32 bf16, MxK): lane m=lane&15, hi=lane>=16.
// us[0..7]  = A[m][kb + hi*8 .. +8)
// us[8..15] = A[m][kb + 16 + hi*8 .. +8)       (per ISA 16-bit A table)
__device__ __forceinline__ Frag load_a_frag(const u16* base, int hi) {
  Frag f;
  f.q[0] = *(const u32x4*)(base + hi * 8);
  f.q[1] = *(const u32x4*)(base + 16 + hi * 8);
  return f;
}

// B-fragment (32x16 bf16, KxN) read from BT[N][K] storage:
// lane n=lane&15 holds K = kb + hi*8 .. +8 and kb + 16 + hi*8 .. +8 (A-table order).
__device__ __forceinline__ Frag load_bt_frag(const u16* base, int hi) {
  Frag f;
  f.q[0] = *(const u32x4*)(base + hi * 8);
  f.q[1] = *(const u32x4*)(base + 16 + hi * 8);
  return f;
}

// ---------------------------------------------------------------------------
// Generic bf16 WMMA GEMM: C[M,N] = A[M,K] * BT[N,K]^T.
// Block = 128 threads (4 waves), block tile 128x64; each wave owns a
// 32x64 sub-tile (8 accumulators) -> 8 WMMAs per k-step against 12 16B
// fragment loads (B fragments amortized over two row groups).
// EPI 0: fp32 +bias      EPI 1: bf16 gelu(+bias)
// EPI 2: q/k head-split bf16 [b][h][t][c] (*scale)
// EPI 3: v head-split bf16 transposed [b][h][c][t]
// EPI 4: fp32 +bias+residual
// ---------------------------------------------------------------------------
template <int EPI>
__global__ __launch_bounds__(128) void gemm_kernel(
    const u16* __restrict__ A, const u16* __restrict__ BT,
    void* __restrict__ dst, const float* __restrict__ bias,
    const float* __restrict__ res, int M, int N, int K, float scale) {
  const int lane = threadIdx.x & 31;
  const int wave = threadIdx.x >> 5;          // 0..3
  const int hi   = lane >> 4;
  const int m16  = lane & 15;
  const int rowBase = blockIdx.y * 128 + wave * 32;
  const int colBase = blockIdx.x * 64;

  v8f acc[2][4];
#pragma unroll
  for (int rg = 0; rg < 2; rg++)
#pragma unroll
    for (int t = 0; t < 4; t++)
#pragma unroll
      for (int i = 0; i < 8; i++) acc[rg][t][i] = 0.0f;

  const u16* arow0 = A + (size_t)(rowBase + m16) * K;
  const u16* arow1 = A + (size_t)(rowBase + 16 + m16) * K;
  for (int kb = 0; kb < K; kb += 32) {
    Frag a0 = load_a_frag(arow0 + kb, hi);
    Frag a1 = load_a_frag(arow1 + kb, hi);
    if (kb + 32 < K) {                        // stream-ahead on the A operand
      __builtin_prefetch(arow0 + kb + 32, 0, 1);
      __builtin_prefetch(arow1 + kb + 32, 0, 1);
    }
#pragma unroll
    for (int nt = 0; nt < 4; nt++) {
      const u16* bptr = BT + (size_t)(colBase + nt * 16 + m16) * K + kb;
      Frag bf = load_bt_frag(bptr, hi);
      acc[0][nt] = __builtin_amdgcn_wmma_f32_16x16x32_bf16(
          false, a0.v, false, bf.v, (short)0, acc[0][nt], false, false);
      acc[1][nt] = __builtin_amdgcn_wmma_f32_16x16x32_bf16(
          false, a1.v, false, bf.v, (short)0, acc[1][nt], false, false);
    }
  }

#pragma unroll
  for (int rg = 0; rg < 2; rg++) {
#pragma unroll
    for (int nt = 0; nt < 4; nt++) {
#pragma unroll
      for (int g = 0; g < 8; g++) {
        const int row = rowBase + rg * 16 + g + hi * 8;   // C layout: VGPR g -> M
        const int col = colBase + nt * 16 + m16;
        float v = acc[rg][nt][g];
        if (EPI == 0 || EPI == 4) {
          v += bias[col];
          if (EPI == 4) v += res[(size_t)row * N + col];
          ((float*)dst)[(size_t)row * N + col] = v;
        } else if (EPI == 1) {
          v = gelu_tanh(v + bias[col]);
          ((u16*)dst)[(size_t)row * N + col] = f2bf(v);
        } else if (EPI == 2) {
          const int b = row >> 11, t = row & 2047;   // M = B*T, T=2048
          const int c = col >> 3,  hd = col & 7;     // '(k h)' head-fast split
          ((u16*)dst)[(((size_t)(b * 8 + hd) * 2048) + t) * 512 + c] = f2bf(v * scale);
        } else if (EPI == 3) {
          const int b = row >> 11, t = row & 2047;
          const int c = col >> 3,  hd = col & 7;
          ((u16*)dst)[(((size_t)(b * 8 + hd) * 512) + c) * 2048 + t] = f2bf(v);
        }
      }
    }
  }
}

// ---------------------------------------------------------------------------
// Flash attention: grid (T/16, B*H), 256 threads (8 waves).
// Wave w: computes score subtile S[:,16w:16w+16] each iteration (Bc=128 keys),
// and owns output columns [64w,64w+64). Q fragments are register-resident.
// Online softmax is parallelized across all 256 threads (each owns 8 elements
// of the 16x128 score tile) with two-level LDS reductions.
// q,k: [bh][t][512] bf16 (pre-scaled by 512^-0.25). vT: [bh][512][t] bf16.
// Output: attn[b][t][h*512+c] bf16.
// ---------------------------------------------------------------------------
__global__ __launch_bounds__(256) void attention_kernel(
    const u16* __restrict__ q, const u16* __restrict__ k,
    const u16* __restrict__ vT, u16* __restrict__ attn) {
  constexpr int T = 2048, D = 512;
  const int lane = threadIdx.x & 31;
  const int wave = threadIdx.x >> 5;
  const int hi   = lane >> 4;
  const int m16  = lane & 15;
  const int r_sm = threadIdx.x & 15;          // softmax: row owned by this thread
  const int cg   = threadIdx.x >> 4;          // softmax: 8-wide column group
  const int bh = blockIdx.y;
  const int b = bh >> 3, hd = bh & 7;
  const int qt0 = blockIdx.x * 16;

  const u16* qb = q  + (size_t)bh * T * D;
  const u16* kp = k  + (size_t)bh * T * D;
  const u16* vb = vT + (size_t)bh * D * T;

  __shared__ __align__(16) float Sl[16][132];   // padded: no 16-way bank conflicts
  __shared__ __align__(16) u16   Pl[16][136];   // stride mult of 8 -> b128 LDS loads
  __shared__ float mrow[16], lrow[16], arow_s[16];
  __shared__ float red[256];                    // partial max / partial sum

  if (threadIdx.x < 16) { mrow[threadIdx.x] = -INFINITY; lrow[threadIdx.x] = 0.0f; }

  // Register-resident Q tile: 16 fragments covering d=512 (128 VGPRs/lane).
  Frag qf[16];
  const u16* qrow = qb + (size_t)(qt0 + m16) * D;
#pragma unroll
  for (int f = 0; f < 16; f++) qf[f] = load_a_frag(qrow + f * 32, hi);

  v8f o[4];
#pragma unroll
  for (int t = 0; t < 4; t++)
#pragma unroll
    for (int i = 0; i < 8; i++) o[t][i] = 0.0f;

  __syncthreads();

  for (int kt = 0; kt < 16; kt++) {             // key tiles of 128
    // ---- S = Q * K^T (this wave's 16x16 subtile, full d=512 reduction)
    v8f s;
#pragma unroll
    for (int i = 0; i < 8; i++) s[i] = 0.0f;
    const u16* krow = kp + (size_t)(kt * 128 + wave * 16 + m16) * D;
    if (kt < 15) __builtin_prefetch(krow + (size_t)128 * D, 0, 1);  // next K tile
#pragma unroll
    for (int f = 0; f < 16; f++) {
      Frag bf = load_bt_frag(krow + f * 32, hi);
      s = __builtin_amdgcn_wmma_f32_16x16x32_bf16(
          false, qf[f].v, false, bf.v, (short)0, s, false, false);
    }
#pragma unroll
    for (int g = 0; g < 8; g++) Sl[g + hi * 8][wave * 16 + m16] = s[g];
    __syncthreads();                                            // BAR1

    // ---- online softmax, parallel over all 256 threads
    {
      float pm = -INFINITY;
#pragma unroll
      for (int j = 0; j < 8; j++) pm = fmaxf(pm, Sl[r_sm][cg * 8 + j]);
      red[r_sm + 16 * cg] = pm;
    }
    __syncthreads();                                            // BAR2
    if (threadIdx.x < 16) {
      const int r = threadIdx.x;
      const float mo = mrow[r];
      float mn = mo;
      for (int c2 = 0; c2 < 16; c2++) mn = fmaxf(mn, red[r + 16 * c2]);
      arow_s[r] = __expf(mo - mn);
      mrow[r] = mn;
    }
    __syncthreads();                                            // BAR3
    {
      const float mn = mrow[r_sm];
      float sum = 0.0f;
#pragma unroll
      for (int j = 0; j < 8; j++) {
        float p = __expf(Sl[r_sm][cg * 8 + j] - mn);
        Pl[r_sm][cg * 8 + j] = f2bf(p);
        sum += p;
      }
      red[r_sm + 16 * cg] = sum;
    }
    __syncthreads();                                            // BAR4
    if (threadIdx.x < 16) {
      const int r = threadIdx.x;
      float sum = 0.0f;
      for (int c2 = 0; c2 < 16; c2++) sum += red[r + 16 * c2];
      lrow[r] = lrow[r] * arow_s[r] + sum;
    }
    // (no trailing barrier needed: next iteration's BAR1..BAR3 chain already
    //  orders this iteration's Pl reads against the next Pl writes)

    // ---- rescale O, accumulate P*V for this wave's 64 output columns
    float al8[8];
#pragma unroll
    for (int g = 0; g < 8; g++) al8[g] = arow_s[g + hi * 8];
#pragma unroll
    for (int nt = 0; nt < 4; nt++)
#pragma unroll
      for (int g = 0; g < 8; g++) o[nt][g] *= al8[g];

    if (kt < 15)                                                 // next V tile
      __builtin_prefetch(vb + (size_t)(wave * 64 + m16) * T + kt * 128 + 128, 0, 1);
#pragma unroll
    for (int ks = 0; ks < 4; ks++) {
      Frag pf;
      pf.q[0] = *(const u32x4*)(&Pl[m16][ks * 32 + hi * 8]);
      pf.q[1] = *(const u32x4*)(&Pl[m16][ks * 32 + 16 + hi * 8]);
#pragma unroll
      for (int nt = 0; nt < 4; nt++) {
        const u16* vp = vb + (size_t)(wave * 64 + nt * 16 + m16) * T + kt * 128 + ks * 32;
        Frag bf = load_bt_frag(vp, hi);
        o[nt] = __builtin_amdgcn_wmma_f32_16x16x32_bf16(
            false, pf.v, false, bf.v, (short)0, o[nt], false, false);
      }
    }
    __syncthreads();                                            // protect arow_s/Sl
  }

  if (threadIdx.x < 16) arow_s[threadIdx.x] = 1.0f / lrow[threadIdx.x];
  __syncthreads();
  float il8[8];
#pragma unroll
  for (int g = 0; g < 8; g++) il8[g] = arow_s[g + hi * 8];

#pragma unroll
  for (int nt = 0; nt < 4; nt++) {
#pragma unroll
    for (int g = 0; g < 8; g++) {
      const int t = qt0 + g + hi * 8;
      const int col = hd * 512 + wave * 64 + nt * 16 + m16;   // 'bthk' flatten
      attn[((size_t)(b * 2048 + t)) * 4096 + col] = f2bf(o[nt][g] * il8[g]);
    }
  }
}

// ---------------------------------------------------------------------------
// Elementwise / reduction helpers
// ---------------------------------------------------------------------------
__global__ void convert_bf16_kernel(const float* __restrict__ in, u16* __restrict__ out, int n) {
  int i = blockIdx.x * blockDim.x + threadIdx.x;
  if (i < n) out[i] = f2bf(in[i]);
}

// in: fp32 [K x N] row-major  ->  out: bf16 [N x K] row-major (transposed)
__global__ void transpose_bf16_kernel(const float* __restrict__ in, u16* __restrict__ out,
                                      int K, int N) {
  int i = blockIdx.x * blockDim.x + threadIdx.x;
  if (i < K * N) {
    int n = i / K, kk = i - n * K;
    out[i] = f2bf(in[(size_t)kk * N + n]);
  }
}

// Deterministic per-batch sum / sumsq (no float atomics; graph replays stable)
__global__ __launch_bounds__(256) void reduce1_kernel(const float* __restrict__ in,
                                                      float* __restrict__ part, int perBatch) {
  __shared__ float s1[256], s2[256];
  const int b = blockIdx.y;
  const int chunk = perBatch / gridDim.x;
  const int iters = chunk / 256;
  int base = b * perBatch + blockIdx.x * chunk + threadIdx.x;
  float a = 0.0f, c = 0.0f;
  for (int i = 0; i < iters; i++) {
    float v = in[base + i * 256];
    a += v; c += v * v;
  }
  s1[threadIdx.x] = a; s2[threadIdx.x] = c;
  __syncthreads();
  for (int off = 128; off > 0; off >>= 1) {
    if (threadIdx.x < off) {
      s1[threadIdx.x] += s1[threadIdx.x + off];
      s2[threadIdx.x] += s2[threadIdx.x + off];
    }
    __syncthreads();
  }
  if (threadIdx.x == 0) {
    part[(b * gridDim.x + blockIdx.x) * 2 + 0] = s1[0];
    part[(b * gridDim.x + blockIdx.x) * 2 + 1] = s2[0];
  }
}

__global__ void reduce2_kernel(const float* __restrict__ part, float* __restrict__ st) {
  __shared__ float s1[128], s2[128];
  const int b = blockIdx.x;
  s1[threadIdx.x] = part[(b * 128 + threadIdx.x) * 2 + 0];
  s2[threadIdx.x] = part[(b * 128 + threadIdx.x) * 2 + 1];
  __syncthreads();
  for (int off = 64; off > 0; off >>= 1) {
    if (threadIdx.x < off) {
      s1[threadIdx.x] += s1[threadIdx.x + off];
      s2[threadIdx.x] += s2[threadIdx.x + off];
    }
    __syncthreads();
  }
  if (threadIdx.x == 0) { st[b * 2 + 0] = s1[0]; st[b * 2 + 1] = s2[0]; }
}

// LayerNorm over (T,k) per batch (axis=[-2,-1]); optional residual add after LN.
__global__ void ln_apply_kernel(const float* __restrict__ in, const float* __restrict__ res,
                                const float* __restrict__ gs, const float* __restrict__ go,
                                const float* __restrict__ st, float* __restrict__ outf,
                                u16* __restrict__ outb) {
  const int i = blockIdx.x * blockDim.x + threadIdx.x;   // total = 4 * 2^20 exactly
  const int b = i >> 20;                                  // T*k = 2048*512 = 2^20
  const int c = i & 511;
  const float inv = 1.0f / 1048576.0f;
  const float mean = st[b * 2 + 0] * inv;
  const float var  = st[b * 2 + 1] * inv - mean * mean;
  const float r = rsqrtf(var + 1e-5f);
  float y = gs[c] * (in[i] - mean) * r + go[c];
  if (res)  y += res[i];
  outf[i] = y;
  if (outb) outb[i] = f2bf(y);
}

// ---------------------------------------------------------------------------
extern "C" void kernel_launch(void* const* d_in, const int* in_sizes, int n_in,
                              void* d_out, int out_size, void* d_ws, size_t ws_size,
                              hipStream_t stream) {
  (void)in_sizes; (void)n_in; (void)out_size; (void)ws_size;

  const float* x   = (const float*)d_in[0];
  const float* Wq  = (const float*)d_in[1];
  const float* Wk  = (const float*)d_in[2];
  const float* Wv  = (const float*)d_in[3];
  const float* Wu  = (const float*)d_in[4];
  const float* bu  = (const float*)d_in[5];
  const float* g1s = (const float*)d_in[6];
  const float* g1o = (const float*)d_in[7];
  const float* W1  = (const float*)d_in[8];
  const float* b1  = (const float*)d_in[9];
  const float* W2  = (const float*)d_in[10];
  const float* b2  = (const float*)d_in[11];
  const float* g2s = (const float*)d_in[12];
  const float* g2o = (const float*)d_in[13];

  // Bump allocator over workspace (~300 MB total)
  char* p = (char*)d_ws;
  auto alloc = [&](size_t bytes) { char* r = p; p += (bytes + 255) & ~(size_t)255; return r; };

  u16*   xb   = (u16*)alloc(8192ull * 512 * 2);
  u16*   WqT  = (u16*)alloc(4096ull * 512 * 2);
  u16*   WkT  = (u16*)alloc(4096ull * 512 * 2);
  u16*   WvT  = (u16*)alloc(4096ull * 512 * 2);
  u16*   WuT  = (u16*)alloc(512ull * 4096 * 2);
  u16*   W1T  = (u16*)alloc(2048ull * 512 * 2);
  u16*   W2T  = (u16*)alloc(512ull * 2048 * 2);
  float* st   = (float*)alloc(64);
  float* pr1  = (float*)alloc(4ull * 128 * 2 * 4);
  float* pr2  = (float*)alloc(4ull * 128 * 2 * 4);
  char*  regB = alloc(3ull * 32 * 2048 * 512 * 2);        // q_r | k_r | v_T (201 MB)
  u16*   attn = (u16*)alloc(8192ull * 4096 * 2);          // 67 MB

  u16* qr = (u16*)regB;
  u16* kr = qr + 32ull * 2048 * 512;
  u16* vT = kr + 32ull * 2048 * 512;
  // After attention, regB is recycled for the post-attention fp32 pipeline:
  float* Y1  = (float*)regB;                                            // 16.8 MB
  float* X1  = (float*)(regB + 16777216ull);                            // 16.8 MB
  u16*   X1b = (u16*)  (regB + 2ull * 16777216ull);                     //  8.4 MB
  u16*   H   = (u16*)  (regB + 2ull * 16777216ull + 8388608ull);        // 33.6 MB
  float* Z   = (float*)(regB + 2ull * 16777216ull + 8388608ull + 33554432ull);

  const float qscale = 0.21022410381342864f;   // 512^(-1/4)

  // 1. precision/layout prep
  convert_bf16_kernel<<<(8192 * 512) / 256, 256, 0, stream>>>(x, xb, 8192 * 512);
  transpose_bf16_kernel<<<(512 * 4096) / 256, 256, 0, stream>>>(Wq, WqT, 512, 4096);
  transpose_bf16_kernel<<<(512 * 4096) / 256, 256, 0, stream>>>(Wk, WkT, 512, 4096);
  transpose_bf16_kernel<<<(512 * 4096) / 256, 256, 0, stream>>>(Wv, WvT, 512, 4096);
  transpose_bf16_kernel<<<(4096 * 512) / 256, 256, 0, stream>>>(Wu, WuT, 4096, 512);
  transpose_bf16_kernel<<<(512 * 2048) / 256, 256, 0, stream>>>(W1, W1T, 512, 2048);
  transpose_bf16_kernel<<<(2048 * 512) / 256, 256, 0, stream>>>(W2, W2T, 2048, 512);

  // 2. QKV projections (head-split epilogues; V stored transposed)
  gemm_kernel<2><<<dim3(64, 64), 128, 0, stream>>>(xb, WqT, qr, nullptr, nullptr, 8192, 4096, 512, qscale);
  gemm_kernel<2><<<dim3(64, 64), 128, 0, stream>>>(xb, WkT, kr, nullptr, nullptr, 8192, 4096, 512, qscale);
  gemm_kernel<3><<<dim3(64, 64), 128, 0, stream>>>(xb, WvT, vT, nullptr, nullptr, 8192, 4096, 512, 1.0f);

  // 3. flash attention -> attn[b][t][h*512+c] bf16
  attention_kernel<<<dim3(128, 32), 256, 0, stream>>>(qr, kr, vT, attn);

  // 4. output projection + LN1 (+ residual x)
  gemm_kernel<0><<<dim3(8, 64), 128, 0, stream>>>(attn, WuT, Y1, bu, nullptr, 8192, 512, 4096, 1.0f);
  reduce1_kernel<<<dim3(128, 4), 256, 0, stream>>>(Y1, pr1, 1 << 20);
  reduce2_kernel<<<4, 128, 0, stream>>>(pr1, st + 0);
  ln_apply_kernel<<<16384, 256, 0, stream>>>(Y1, x, g1s, g1o, st + 0, X1, X1b);

  // 5. FFN + LN2
  gemm_kernel<1><<<dim3(32, 64), 128, 0, stream>>>(X1b, W1T, H, b1, nullptr, 8192, 2048, 512, 1.0f);
  gemm_kernel<4><<<dim3(8, 64), 128, 0, stream>>>(H, W2T, Z, b2, X1, 8192, 512, 2048, 1.0f);
  reduce1_kernel<<<dim3(128, 4), 256, 0, stream>>>(Z, pr2, 1 << 20);
  reduce2_kernel<<<4, 128, 0, stream>>>(pr2, st + 8);
  ln_apply_kernel<<<16384, 256, 0, stream>>>(Z, nullptr, g2s, g2o, st + 8, (float*)d_out, nullptr);
}
